// rnn_pack_encoder_68161130987651
// MI455X (gfx1250) — matrix operand order repaired and checked by reading, not compiled, not run on hardware
//
#include <hip/hip_runtime.h>
#include <hip/hip_bf16.h>

#define BB 32
#define TT 2048
#define DD 256
#define HH 256
#define G3 768   // 3*H

typedef __bf16 v8bf  __attribute__((ext_vector_type(8)));
typedef __bf16 v16bf __attribute__((ext_vector_type(16)));
typedef float  v8f   __attribute__((ext_vector_type(8)));

static __device__ __forceinline__ v16bf cat16(v8bf lo, v8bf hi) {
    return __builtin_shufflevector(lo, hi, 0,1,2,3,4,5,6,7,8,9,10,11,12,13,14,15);
}

// ---------------------------------------------------------------- utilities
__global__ void zero_f32_kernel(float* __restrict__ p, size_t n) {
    size_t i = (size_t)blockIdx.x * blockDim.x + threadIdx.x;
    size_t stride = (size_t)gridDim.x * blockDim.x;
    for (; i < n; i += stride) p[i] = 0.0f;
}

__global__ void cvt_bf16_kernel(const float* __restrict__ s, __bf16* __restrict__ d, size_t n) {
    size_t i = (size_t)blockIdx.x * blockDim.x + threadIdx.x;
    size_t stride = (size_t)gridDim.x * blockDim.x;
    for (; i < n; i += stride) d[i] = (__bf16)s[i];
}

// ---------------------------------------------------------------- phase 1:
// GI[m,n] = sum_k Xb[m,k] * Wih[n,k] + b_ih[n];  M=B*T=65536, N=768, K=256
// 32x32 macro-tile per wave (2x2 WMMA tiles): 4 WMMAs per k-step from 2 A + 2 B
// tile loads -> 2x the arithmetic intensity of a 16x16/wave mapping.
__global__ __launch_bounds__(256)
void gi_gemm_kernel(const __bf16* __restrict__ Xb,
                    const __bf16* __restrict__ Wb,
                    const float*  __restrict__ bih,
                    float* __restrict__ GI) {
    const int wv   = threadIdx.x >> 5;
    const int lane = threadIdx.x & 31;
    const int wg   = blockIdx.x * 8 + wv;           // global wave id
    const int nt2  = wg % 24;                       // 24 N macro-tiles of 32 -> 768
    const int mt2  = wg / 24;                       // 2048 M macro-tiles of 32
    const int m0   = mt2 * 32;
    const int n0   = nt2 * 32;
    const int hsel = lane >> 4;                     // lane half select
    const int lm   = lane & 15;

    v8f acc00 = {}, acc01 = {}, acc10 = {}, acc11 = {};
    const __bf16* arow0 = Xb + (size_t)(m0 + lm) * DD;        // rows m0..m0+15
    const __bf16* arow1 = arow0 + (size_t)16 * DD;            // rows m0+16..m0+31
    const __bf16* brow0 = Wb + (size_t)(n0 + lm) * DD;        // cols n0..n0+15
    const __bf16* brow1 = brow0 + (size_t)16 * DD;            // cols n0+16..n0+31
#pragma unroll
    for (int kk = 0; kk < 8; ++kk) {                // K = 8 * 32
        // A 16x32 bf16: lanes0-15 K{0..7,16..23}; lanes16-31 K{8..15,24..31}
        const __bf16* ap0 = arow0 + kk * 32 + hsel * 8;
        const __bf16* ap1 = arow1 + kk * 32 + hsel * 8;
        v16bf A0 = cat16(*(const v8bf*)ap0, *(const v8bf*)(ap0 + 16));
        v16bf A1 = cat16(*(const v8bf*)ap1, *(const v8bf*)(ap1 + 16));
        // B 32x16 bf16: lanes0-15 K{0..7,16..23}; lanes16-31 K{8..15,24..31}
        const __bf16* bp0 = brow0 + kk * 32 + hsel * 8;
        const __bf16* bp1 = brow1 + kk * 32 + hsel * 8;
        v16bf B0 = cat16(*(const v8bf*)bp0, *(const v8bf*)(bp0 + 16));
        v16bf B1 = cat16(*(const v8bf*)bp1, *(const v8bf*)(bp1 + 16));
        acc00 = __builtin_amdgcn_wmma_f32_16x16x32_bf16(false, A0, false, B0, (short)0, acc00, false, false);
        acc01 = __builtin_amdgcn_wmma_f32_16x16x32_bf16(false, A0, false, B1, (short)0, acc01, false, false);
        acc10 = __builtin_amdgcn_wmma_f32_16x16x32_bf16(false, A1, false, B0, (short)0, acc10, false, false);
        acc11 = __builtin_amdgcn_wmma_f32_16x16x32_bf16(false, A1, false, B1, (short)0, acc11, false, false);
    }
    const float bias0 = bih[n0 + lm];
    const float bias1 = bih[n0 + 16 + lm];
#pragma unroll
    for (int r = 0; r < 8; ++r) {                   // C/D: M = r + 8*hsel, N = lm
        const int mmA = m0 + r + hsel * 8;
        const int mmB = mmA + 16;
        GI[(size_t)mmA * G3 + n0 + lm]       = acc00[r] + bias0;
        GI[(size_t)mmA * G3 + n0 + 16 + lm]  = acc01[r] + bias1;
        GI[(size_t)mmB * G3 + n0 + lm]       = acc10[r] + bias0;
        GI[(size_t)mmB * G3 + n0 + 16 + lm]  = acc11[r] + bias1;
    }
}

// ---------------------------------------------------------------- phase 2:
// Sequential recurrence. 2 blocks (16 batch rows each) x 768 threads (24 waves).
// W_hh(bf16) lives in VGPRs (2 N-tiles * 8 K-tiles = 128 VGPRs/wave).
// LDS (dynamic, 104 KB): h bf16 (8KB) | gh f32 (48KB) | GI stage f32 (48KB).
// The per-step GI slab is pulled LDS-ward with GLOBAL_LOAD_ASYNC_TO_LDS_B128
// overlapping the WMMA k-loop; GI for t+8 is prefetched into L2.
__global__ __launch_bounds__(768)
void gru_rec_kernel(const float*  __restrict__ GI,
                    const __bf16* __restrict__ Whhb,
                    const float*  __restrict__ bhhg,
                    const int*    __restrict__ seg,
                    const int*    __restrict__ len,
                    float* __restrict__ out) {
    extern __shared__ char smem[];
    __bf16* h_bf   = (__bf16*)smem;                       // 16*256 bf16 =  8 KB
    float*  gh     = (float*)(smem + 8192);               // 16*768 f32  = 48 KB
    float*  gi_lds = (float*)(smem + 8192 + 49152);       // 16*768 f32  = 48 KB

    const int tid  = threadIdx.x;
    const int wv   = tid >> 5;            // 0..23
    const int lane = tid & 31;
    const int hsel = lane >> 4;
    const int lm   = lane & 15;
    const int mrow0 = blockIdx.x * 16;    // batch-row slab base

    // ---- preload W_hh tiles for this wave into registers (constant over t)
    v16bf Breg[2][8];
    float bhh[2];
#pragma unroll
    for (int j = 0; j < 2; ++j) {
        const int n = wv * 32 + j * 16 + lm;
        bhh[j] = bhhg[n];
        const __bf16* brow = Whhb + (size_t)n * HH;
#pragma unroll
        for (int kk = 0; kk < 8; ++kk) {
            const __bf16* bp = brow + kk * 32 + hsel * 8;
            Breg[j][kk] = cat16(*(const v8bf*)bp, *(const v8bf*)(bp + 16));
        }
    }

    // ---- init hidden state + per-thread packing counters
    for (int i = tid; i < 16 * HH; i += 768) h_bf[i] = (__bf16)0.0f;
    int idxloc[6];
#pragma unroll
    for (int c = 0; c < 6; ++c) idxloc[c] = 0;
    __syncthreads();

    for (int t = 0; t < TT; ++t) {
        // -- kick off async DMA of this step's GI slab (16 rows x 768 f32) to LDS.
        //    3072 16B chunks over 768 threads = 4 chunks each; tracked by ASYNCcnt.
#pragma unroll
        for (int k = 0; k < 4; ++k) {
            const int q = tid + k * 768;               // chunk id
            const int m = q / 192;                     // 192 chunks per row
            const int c = (q % 192) * 4;               // float offset in row
            const unsigned long long ga =
                (unsigned long long)(GI + ((size_t)(mrow0 + m) * TT + t) * G3 + c);
            const unsigned ldsoff = (unsigned)(unsigned long long)(gi_lds + q * 4);
            asm volatile("global_load_async_to_lds_b128 %0, %1, off"
                         :: "v"(ldsoff), "v"(ga) : "memory");
        }
        // -- L2 prefetch of the slab 8 steps ahead (one 64B line per thread)
        if (t + 8 < TT) {
            const int m = tid / 48;                    // 48 lines per 3072B row
            const int c = (tid % 48) * 64;
            __builtin_prefetch((const char*)(GI + ((size_t)(mrow0 + m) * TT + (t + 8)) * G3) + c, 0, 1);
        }

        // -- gh = h @ W_hh^T  (A tiles streamed from LDS, overlaps the DMA)
        v8f acc0 = {}, acc1 = {};
#pragma unroll
        for (int kk = 0; kk < 8; ++kk) {
            const __bf16* ap = &h_bf[lm * HH + kk * 32 + hsel * 8];
            v16bf A = cat16(*(const v8bf*)ap, *(const v8bf*)(ap + 16));
            acc0 = __builtin_amdgcn_wmma_f32_16x16x32_bf16(false, A, false, Breg[0][kk],
                                                           (short)0, acc0, false, false);
            acc1 = __builtin_amdgcn_wmma_f32_16x16x32_bf16(false, A, false, Breg[1][kk],
                                                           (short)0, acc1, false, false);
        }
#pragma unroll
        for (int r = 0; r < 8; ++r) {
            const int mm = r + hsel * 8;
            gh[mm * G3 + wv * 32 + lm]      = acc0[r] + bhh[0];
            gh[mm * G3 + wv * 32 + 16 + lm] = acc1[r] + bhh[1];
        }
        // own async chunks landed in LDS; barrier makes everyone's visible
        asm volatile("s_wait_asynccnt 0x0" ::: "memory");
        __syncthreads();

        // -- elementwise gate math + segment reset + packed write (all LDS reads)
        int slot = 0;
        for (int i = tid; i < 16 * HH; i += 768, ++slot) {
            const int m = i >> 8;            // H == 256
            const int j = i & 255;
            const int b = mrow0 + m;
            const float gr  = gi_lds[m * G3 + j]       + gh[m * G3 + j];
            const float gz  = gi_lds[m * G3 + 256 + j] + gh[m * G3 + 256 + j];
            const float gni = gi_lds[m * G3 + 512 + j];
            const float gnh = gh[m * G3 + 512 + j];
            const float r  = 1.0f / (1.0f + __expf(-gr));
            const float z  = 1.0f / (1.0f + __expf(-gz));
            const float nn = tanhf(gni + r * gnh);
            const float hold = (float)h_bf[m * HH + j];
            const float hnew = (1.0f - z) * nn + z * hold;
            const int sg = seg[(size_t)b * TT + t];
            const bool wr = (sg == 0) && (t < len[b]);
            if (wr) out[((size_t)b * TT + idxloc[slot]) * HH + j] = hnew;
            idxloc[slot] += wr ? 1 : 0;
            h_bf[m * HH + j] = (__bf16)(sg ? hnew : 0.0f);
        }
        __syncthreads();
    }
}

// ---------------------------------------------------------------- final idx
__global__ void idx_final_kernel(const int* __restrict__ seg,
                                 const int* __restrict__ len,
                                 int* __restrict__ idx_out) {
    const int b = threadIdx.x;
    if (b < BB) {
        const int L = len[b];
        int c = 0;
        for (int t = 0; t < TT; ++t)
            c += ((seg[(size_t)b * TT + t] == 0) && (t < L)) ? 1 : 0;
        idx_out[b] = c;
    }
}

// ---------------------------------------------------------------- launch
extern "C" void kernel_launch(void* const* d_in, const int* in_sizes, int n_in,
                              void* d_out, int out_size, void* d_ws, size_t ws_size,
                              hipStream_t stream) {
    const float* x    = (const float*)d_in[0];   // [B,T,D]
    const int*   seg  = (const int*)  d_in[1];   // [B,T]
    const int*   len  = (const int*)  d_in[2];   // [B]
    const float* Wih  = (const float*)d_in[3];   // [3H,D]
    const float* Whh  = (const float*)d_in[4];   // [3H,H]
    const float* bih  = (const float*)d_in[5];   // [3H]
    const float* bhh  = (const float*)d_in[6];   // [3H]

    float* out = (float*)d_out;                                // [B,T,H] fp32
    int*   idx_out = (int*)(out + (size_t)BB * TT * HH);       // [B] int32 tail

    // workspace layout
    char* w = (char*)d_ws;
    float*  GI   = (float*)w;                                  // 65536*768 f32 (192 MiB)
    size_t off   = (size_t)BB * TT * G3 * sizeof(float);
    __bf16* Xb   = (__bf16*)(w + off);  off += (size_t)BB * TT * DD * 2;
    __bf16* Wihb = (__bf16*)(w + off);  off += (size_t)G3 * DD * 2;
    __bf16* Whhb = (__bf16*)(w + off);

    // 1) zero the packed-output buffer (reference zero-inits; harness poisons)
    zero_f32_kernel<<<2048, 256, 0, stream>>>(out, (size_t)BB * TT * HH);

    // 2) fp32 -> bf16 conversions
    cvt_bf16_kernel<<<4096, 256, 0, stream>>>(x,   Xb,   (size_t)BB * TT * DD);
    cvt_bf16_kernel<<<192,  256, 0, stream>>>(Wih, Wihb, (size_t)G3 * DD);
    cvt_bf16_kernel<<<192,  256, 0, stream>>>(Whh, Whhb, (size_t)G3 * HH);

    // 3) bulk input projection: GI = X * W_ih^T + b_ih  (WMMA bf16, 32x32/wave)
    //    macro-tiles = (65536/32) * (768/32) = 49152 waves; 8 waves per block
    gi_gemm_kernel<<<6144, 256, 0, stream>>>(Xb, Wihb, bih, GI);

    // 4) sequential segmented GRU recurrence (WMMA bf16 + async GI staging)
    gru_rec_kernel<<<2, 768, 104 * 1024, stream>>>(GI, Whhb, bhh, seg, len, out);

    // 5) final packing counters
    idx_final_kernel<<<1, 32, 0, stream>>>(seg, len, idx_out);

    (void)in_sizes; (void)n_in; (void)out_size; (void)ws_size;
}
